// Generator_10926396801165
// MI455X (gfx1250) — matrix operand order repaired and checked by reading, not compiled, not run on hardware
//
#include <hip/hip_runtime.h>
#include <stdint.h>

// ---------------- problem constants ----------------
#define H      106
#define HP     112          // hidden padded to 7*16
#define IN0    100
#define S      512
#define B      256
#define NLAY   11
#define GATES  4
#define NP     (GATES*HP)   // 448 padded gate dim
#define KTOT   (2*HP)       // 224 combined K (x | h)
// ---------------- tiling / launch ----------------
#define WSTRIDE 232         // LDS W row stride (bf16 units): 464B, 16B aligned, odd-ish banks
#define HSTRIDE 120         // LDS h row stride (bf16 units): 240B, 16B aligned
#define BSLICES 4
#define MROWS   64          // batch rows per slice
#define HTILES  7
#define WAVES   28          // 4 mtiles * 7 htiles -> one (mt,ht) task per wave
#define TPB     (WAVES*32)  // 896 threads
#define RING    32          // ring slots per inter-layer stage

typedef unsigned short u16;
typedef __attribute__((ext_vector_type(8)))  u16    v8u;
typedef __attribute__((ext_vector_type(16))) u16    v16u;
typedef __attribute__((ext_vector_type(16))) __bf16 v16bf;
typedef __attribute__((ext_vector_type(8)))  float  v8f;

static __device__ __forceinline__ u16 f2bf(float f) {
    unsigned int u = __float_as_uint(f);
    u += 0x7FFFu + ((u >> 16) & 1u);     // round-to-nearest-even
    return (u16)(u >> 16);
}
static __device__ __forceinline__ float sigf(float x) {
    return 1.0f / (1.0f + __expf(-x));
}
static __device__ __forceinline__ float tanh_fast(float x) {
    x = fminf(fmaxf(x, -15.0f), 15.0f);
    float e = __expf(-2.0f * x);
    return (1.0f - e) / (1.0f + e);
}

// ---------------- stage0: f32 noise -> bf16, K padded to 112 ----------------
__global__ void noise_to_bf16(const float* __restrict__ noise, u16* __restrict__ stage0) {
    size_t i = (size_t)blockIdx.x * blockDim.x + threadIdx.x;
    size_t total = (size_t)S * B * HP;
    if (i >= total) return;
    int k = (int)(i % HP);
    size_t row = i / HP;
    float v = (k < IN0) ? noise[row * IN0 + k] : 0.0f;
    stage0[i] = f2bf(v);
}

// ---------------- persistent layer-pipeline LSTM ----------------
__global__ __launch_bounds__(TPB, 1) void lstm_pipeline(
    const float* __restrict__ Wih0, const float* __restrict__ Whh0,
    const float* __restrict__ bih0, const float* __restrict__ bhh0,
    const float* __restrict__ WihR, const float* __restrict__ WhhR,
    const float* __restrict__ bihR, const float* __restrict__ bhhR,
    u16* __restrict__ stage0, u16* __restrict__ rings,
    int* prod, int* cons, float* __restrict__ out)
{
    extern __shared__ u16 smem[];
    u16*   Wl    = smem;                               // [NP][WSTRIDE] bf16
    u16*   Hl    = smem + (size_t)NP * WSTRIDE;        // [MROWS][HSTRIDE] bf16
    float* biasl = (float*)(smem + (size_t)NP * WSTRIDE + (size_t)MROWS * HSTRIDE); // [NP]

    const int layer = blockIdx.x / BSLICES;
    const int slice = blockIdx.x % BSLICES;

    const int    Kin = (layer == 0) ? IN0 : H;
    const float* Wih = (layer == 0) ? Wih0 : WihR + (size_t)(layer - 1) * GATES * H * H;
    const float* Whh = (layer == 0) ? Whh0 : WhhR + (size_t)(layer - 1) * GATES * H * H;
    const float* bih = (layer == 0) ? bih0 : bihR + (size_t)(layer - 1) * GATES * H;
    const float* bhh = (layer == 0) ? bhh0 : bhhR + (size_t)(layer - 1) * GATES * H;

    // ---- repack weights (f32 -> bf16) into LDS: row n' = gate*112+h, cols [0,112)=Wih, [112,224)=Whh
    for (int idx = threadIdx.x; idx < NP * KTOT; idx += TPB) {
        int np = idx / KTOT, k = idx % KTOT;
        int gate = np / HP, h = np % HP;
        float v = 0.0f;
        if (h < H) {
            int r = gate * H + h;
            if (k < HP) { if (k < Kin) v = Wih[(size_t)r * Kin + k]; }
            else        { int k2 = k - HP; if (k2 < H) v = Whh[(size_t)r * H + k2]; }
        }
        Wl[(size_t)np * WSTRIDE + k] = f2bf(v);
    }
    for (int idx = threadIdx.x; idx < MROWS * HSTRIDE; idx += TPB) Hl[idx] = 0;
    for (int idx = threadIdx.x; idx < NP; idx += TPB) {
        int gate = idx / HP, h = idx % HP;
        biasl[idx] = (h < H) ? (bih[gate * H + h] + bhh[gate * H + h]) : 0.0f;
    }
    __syncthreads();

    // ---- per-wave fixed tile assignment
    const int wave  = threadIdx.x >> 5;
    const int lane  = threadIdx.x & 31;
    const int mt    = wave / HTILES;        // 0..3  (16 batch rows each)
    const int ht    = wave % HTILES;        // 0..6  (16 hidden cells each)
    const int khalf = lane >> 4;
    const int lcol  = lane & 15;

    const int mrowA = mt * 16 + lcol;       // A-fragment row (this lane)
    const u16* hrow = Hl + (size_t)mrowA * HSTRIDE;

    const u16* wbase[GATES];
    float biasv[GATES];
#pragma unroll
    for (int g = 0; g < GATES; ++g) {
        wbase[g] = Wl + (size_t)(g * HP + ht * 16 + lcol) * WSTRIDE;
        biasv[g] = biasl[g * HP + ht * 16 + lcol];
    }

    const u16* stageIn  = (layer == 0) ? stage0 : rings + (size_t)(layer - 1) * RING * B * HP;
    u16*       stageOut = (layer < NLAY - 1) ? rings + (size_t)layer * RING * B * HP : nullptr;

    v8f cacc = {};                          // persistent cell state tile (f32, in registers)

    for (int t = 0; t < S; ++t) {
        // ---- wait for producer (layer-1) to publish step t of our batch slice
        if (layer > 0) {
            if (threadIdx.x == 0) {
                int* p = &prod[(layer - 1) * BSLICES + slice];
                while (__hip_atomic_load(p, __ATOMIC_ACQUIRE, __HIP_MEMORY_SCOPE_AGENT) < t + 1)
                    __builtin_amdgcn_s_sleep(1);
            }
            __syncthreads();
        }

        const int slot = (layer == 0) ? t : (t % RING);
        const u16* xrow = stageIn + ((size_t)slot * B + (slice * MROWS + mrowA)) * HP;

        // ---- GEMM: [x_t | h_{t-1}] (16x224) @ W^T -> four 16x16 gate tiles
        v8f acc[GATES];
#pragma unroll
        for (int g = 0; g < GATES; ++g) {
            float bv = biasv[g];
            acc[g] = (v8f){bv, bv, bv, bv, bv, bv, bv, bv};
        }
#pragma unroll
        for (int kk = 0; kk < 7; ++kk) {
            const int kbase = kk * 32;
            const int K1 = kbase + khalf * 8;
            const int K2 = K1 + 16;
            v8u r1, r2;
            if (kk <= 3) r1 = *(const v8u*)(xrow + K1);       // x half (global, L2-hot)
            else         r1 = *(const v8u*)(hrow + (K1 - HP)); // h half (LDS)
            if (kk <= 2) r2 = *(const v8u*)(xrow + K2);
            else         r2 = *(const v8u*)(hrow + (K2 - HP));
            v16bf afrag = __builtin_bit_cast(v16bf,
                __builtin_shufflevector(r1, r2, 0,1,2,3,4,5,6,7,8,9,10,11,12,13,14,15));
#pragma unroll
            for (int g = 0; g < GATES; ++g) {
                const u16* bp = wbase[g] + kbase + khalf * 8;
                v8u b1 = *(const v8u*)(bp);
                v8u b2 = *(const v8u*)(bp + 16);
                v16bf bfrag = __builtin_bit_cast(v16bf,
                    __builtin_shufflevector(b1, b2, 0,1,2,3,4,5,6,7,8,9,10,11,12,13,14,15));
                acc[g] = __builtin_amdgcn_wmma_f32_16x16x32_bf16(
                    false, afrag, false, bfrag, (short)0, acc[g], false, false);
            }
        }

        __syncthreads();  // everyone done reading Hl and stage slot t
        if (threadIdx.x == 0) {
            if (layer > 0)
                __hip_atomic_fetch_add(&cons[layer * BSLICES + slice], 1,
                                       __ATOMIC_RELEASE, __HIP_MEMORY_SCOPE_AGENT);
            if (layer < NLAY - 1 && t >= RING) {   // backpressure: ring slot must be drained
                int* c = &cons[(layer + 1) * BSLICES + slice];
                while (__hip_atomic_load(c, __ATOMIC_ACQUIRE, __HIP_MEMORY_SCOPE_AGENT) < t - RING + 1)
                    __builtin_amdgcn_s_sleep(1);
            }
        }
        __syncthreads();

        // ---- elementwise LSTM cell, all in-register (this wave owns i/f/g/o + c for its tile)
        const int hc = ht * 16 + lcol;          // hidden cell (column)
        const bool hvalid = hc < H;
        float ho[8];
#pragma unroll
        for (int v = 0; v < 8; ++v) {
            float iv = sigf(acc[0][v]);
            float fv = sigf(acc[1][v]);
            float gv = tanh_fast(acc[2][v]);
            float ov = sigf(acc[3][v]);
            float c  = fv * cacc[v] + iv * gv;
            cacc[v]  = c;
            ho[v]    = ov * tanh_fast(c);
        }
#pragma unroll
        for (int v = 0; v < 8; ++v) {
            int m    = v + khalf * 8;           // row within tile (C/D layout)
            int mrow = mt * 16 + m;
            int brow = slice * MROWS + mrow;
            float hval = hvalid ? ho[v] : 0.0f;
            u16 hb = f2bf(hval);
            Hl[(size_t)mrow * HSTRIDE + hc] = hb;                       // next-step recurrent input
            if (layer < NLAY - 1) {
                stageOut[((size_t)(t % RING) * B + brow) * HP + hc] = hb;
            } else if (hvalid) {
                out[((size_t)t * B + brow) * H + hc] = hval;
            }
        }

        __threadfence();
        __syncthreads();
        if (threadIdx.x == 0 && layer < NLAY - 1)
            __hip_atomic_fetch_add(&prod[layer * BSLICES + slice], 1,
                                   __ATOMIC_RELEASE, __HIP_MEMORY_SCOPE_AGENT);
    }
}

extern "C" void kernel_launch(void* const* d_in, const int* in_sizes, int n_in,
                              void* d_out, int out_size, void* d_ws, size_t ws_size,
                              hipStream_t stream) {
    (void)in_sizes; (void)n_in; (void)out_size; (void)ws_size;
    const float* noise = (const float*)d_in[0];
    const float* Wih0  = (const float*)d_in[1];
    const float* Whh0  = (const float*)d_in[2];
    const float* bih0  = (const float*)d_in[3];
    const float* bhh0  = (const float*)d_in[4];
    const float* WihR  = (const float*)d_in[5];
    const float* WhhR  = (const float*)d_in[6];
    const float* bihR  = (const float*)d_in[7];
    const float* bhhR  = (const float*)d_in[8];
    float* out = (float*)d_out;

    // d_ws layout: [0,1024) counters | stage0 (full S, bf16) | 10 ring stages (bf16)
    char* ws = (char*)d_ws;
    int*  ctr    = (int*)ws;                 // prod[44] then cons[44] (within 512B)
    int*  prod   = ctr;
    int*  cons   = ctr + 64;
    u16*  stage0 = (u16*)(ws + 1024);
    u16*  rings  = (u16*)(ws + 1024 + (size_t)S * B * HP * 2);

    hipMemsetAsync(ws, 0, 1024, stream);     // reset progress counters every launch

    const size_t total0 = (size_t)S * B * HP;
    noise_to_bf16<<<dim3((unsigned)((total0 + 255) / 256)), dim3(256), 0, stream>>>(noise, stage0);

    const size_t smemBytes = (size_t)NP * WSTRIDE * 2   // weights bf16
                           + (size_t)MROWS * HSTRIDE * 2 // h state bf16
                           + (size_t)NP * 4;             // bias f32
    hipFuncSetAttribute((const void*)lstm_pipeline,
                        hipFuncAttributeMaxDynamicSharedMemorySize, (int)smemBytes);

    lstm_pipeline<<<dim3(NLAY * BSLICES), dim3(TPB), smemBytes, stream>>>(
        Wih0, Whh0, bih0, bhh0, WihR, WhhR, bihR, bhhR,
        stage0, rings, prod, cons, out);
}
